// NNConvProt_27367531610703
// MI455X (gfx1250) — matrix operand order repaired and checked by reading, not compiled, not run on hardware
//
#include <hip/hip_runtime.h>

typedef __attribute__((ext_vector_type(16))) _Float16 v16h;
typedef __attribute__((ext_vector_type(8)))  float    v8f;

#define GN_NODES  65536
#define GN_EDGES  262144
#define GN_GRAPHS 2048
#define GNF 16
#define GEF 6

// ---------------- monotonic float<->uint key for atomicMax on floats ----------------
__device__ __forceinline__ unsigned f2key(float f) {
  unsigned u = __float_as_uint(f);
  return (u & 0x80000000u) ? ~u : (u | 0x80000000u);
}
__device__ __forceinline__ float key2f(unsigned u) {
  return (u & 0x80000000u) ? __uint_as_float(u & 0x7fffffffu) : __uint_as_float(~u);
}

// ---------------- workspace zero-init (ws is poisoned by harness) ----------------
__global__ void zero_kernel(float* __restrict__ p, int n) {
  const int stride = gridDim.x * blockDim.x;
  for (int i = blockIdx.x * blockDim.x + threadIdx.x; i < n; i += stride) p[i] = 0.0f;
}

// ---------------- attention gate, pass 1: g[n], per-graph max ----------------
__global__ void gate1_kernel(const float* __restrict__ x,
                             const float* __restrict__ gw1, const float* __restrict__ gb1,
                             const float* __restrict__ gw2, const float* __restrict__ gb2,
                             const int* __restrict__ batch,
                             float* __restrict__ gbuf, unsigned* __restrict__ gmax) {
  const int n = blockIdx.x * blockDim.x + threadIdx.x;
  if (n >= GN_NODES) return;
  float xr[GNF];
#pragma unroll
  for (int i = 0; i < GNF; ++i) xr[i] = x[n * GNF + i];
  float g = gb2[0];
#pragma unroll
  for (int j = 0; j < GNF; ++j) {
    float t = gb1[j];
#pragma unroll
    for (int i = 0; i < GNF; ++i) t = fmaf(xr[i], gw1[i * GNF + j], t);
    g = fmaf(fmaxf(t, 0.0f), gw2[j], g);
  }
  gbuf[n] = g;
  atomicMax(&gmax[batch[n]], f2key(g));
}

// ---------------- attention gate, pass 2: exp-sum and weighted numerator ----------------
__global__ void gate2_kernel(const float* __restrict__ x, const int* __restrict__ batch,
                             const float* __restrict__ gbuf, const unsigned* __restrict__ gmax,
                             float* __restrict__ gsum, float* __restrict__ gnum) {
  const int n = blockIdx.x * blockDim.x + threadIdx.x;
  if (n >= GN_NODES) return;
  const int b = batch[n];
  const float e = __expf(gbuf[n] - key2f(gmax[b]));
  atomicAdd(&gsum[b], e);
#pragma unroll
  for (int i = 0; i < GNF; ++i) atomicAdd(&gnum[b * GNF + i], e * x[n * GNF + i]);
}

// ---------------- fused NNConv edge kernel: msg = (h ⊗ x_src) @ W2' (+bias rows), WMMA ----------------
// Each wave processes tiles of 16 edges. The static reshaped weight W2'[(k*IN+i), o]
// (plus IN bias rows of b2, zero-padded to KS*32) is staged once per wave directly in
// the v_wmma_f32_16x16x32_f16 B-register layout and reused across the edge loop.
template <int IN, int OUT>
__global__ void __launch_bounds__(256)
nnconv_edge_kernel(const float* __restrict__ x,
                   const float* __restrict__ ea,
                   const int* __restrict__ src,
                   const int* __restrict__ dst,
                   const float* __restrict__ w1, const float* __restrict__ b1,
                   const float* __restrict__ w2, const float* __restrict__ b2,
                   float* __restrict__ agg) {
  constexpr int KTOT = 16 * IN + IN;        // 272 (conv1) / 544 (conv2)
  constexpr int KS   = (KTOT + 31) / 32;    // 9 / 17  K-steps of 32
  constexpr int NT   = OUT / 16;            // 2 / 1   N-tiles of 16

  __shared__ float sW1[GEF * 16];
  __shared__ float sB1[16];
  if (threadIdx.x < GEF * 16) sW1[threadIdx.x] = w1[threadIdx.x];
  if (threadIdx.x < 16)       sB1[threadIdx.x] = b1[threadIdx.x];
  __syncthreads();

  const int lane = threadIdx.x & 31;
  const int half = lane >> 4;   // which K-half of the A layout this lane holds
  const int arow = lane & 15;   // A-matrix row (edge within tile)
  const int wave = (blockIdx.x * blockDim.x + threadIdx.x) >> 5;
  const int nwav = (gridDim.x * blockDim.x) >> 5;

  // B layout for 16x16x32 f16: lane = N column (t*16+lane%16), element j = K row (s*32 + table order of j, half).
  v16h B[KS][NT];
#pragma unroll
  for (int s = 0; s < KS; ++s) {
#pragma unroll
    for (int t = 0; t < NT; ++t) {
#pragma unroll
      for (int j = 0; j < 16; ++j) {
        const int kk = s * 32 + ((j < 8) ? (8 * half + j) : (16 + 8 * half + (j - 8)));
        const int o = t * 16 + arow;
        float v = 0.0f;
        if (kk < 16 * IN) {
          const int k = kk / IN;
          const int i = kk % IN;
          v = w2[k * (IN * OUT) + i * OUT + o];
        } else if (kk < 16 * IN + IN) {
          v = b2[(kk - 16 * IN) * OUT + o];
        }
        B[s][t][j] = (_Float16)v;
      }
    }
  }

  const int ntiles = GN_EDGES / 16;
  for (int tile = wave; tile < ntiles; tile += nwav) {
    const int e0 = tile * 16;
    const int e  = e0 + arow;               // lanes l and l+16 share an edge row

    // edge MLP hidden: hh[k] = relu(ea @ W1 + b1)   (f16 for the outer product)
    float a[GEF];
#pragma unroll
    for (int j = 0; j < GEF; ++j) a[j] = ea[e * GEF + j];
    _Float16 hh[16];
#pragma unroll
    for (int k = 0; k < 16; ++k) {
      float t = sB1[k];
#pragma unroll
      for (int j = 0; j < GEF; ++j) t = fmaf(a[j], sW1[j * 16 + k], t);
      hh[k] = (_Float16)fmaxf(t, 0.0f);
    }

    // gather source-node features (x is L2 resident)
    float xr[IN];
    const float* xp = x + (size_t)src[e] * IN;
#pragma unroll
    for (int i = 0; i < IN; ++i) xr[i] = xp[i];

    // per-lane x slice matching A layout K positions (select on `half`, constant indices)
    _Float16 xs[16];
#pragma unroll
    for (int j = 0; j < 16; ++j) {
      if constexpr (IN == 16) {
        const int q = j & 7;                         // i = half*8 + (j&7)
        xs[j] = (_Float16)(half ? xr[8 + q] : xr[q]);
      } else {
        const int i0 = (j < 8) ? j : (j + 8);        // i = i0 + half*8
        xs[j] = (_Float16)(half ? xr[i0 + 8] : xr[i0]);
      }
    }

    v8f zero8 = {};
    v8f acc[NT];
#pragma unroll
    for (int t = 0; t < NT; ++t) acc[t] = zero8;

#pragma unroll
    for (int s = 0; s < KS; ++s) {
      // A element j of lane (arow, half) holds K = s*32 + kloc(j, half):
      //   IN==16: K = 16k+i with k = 2s + (j>=8), i = half*8 + (j&7)
      //   IN==32: K = 32k+i with k = s,           i covered by xs[j]
      v16h A;
#pragma unroll
      for (int j = 0; j < 16; ++j) {
        _Float16 v;
        if constexpr (IN == 16) {
          const int k = 2 * s + ((j >= 8) ? 1 : 0);
          if (k < 16)       v = hh[k] * xs[j];   // outer-product rows
          else if (k == 16) v = xs[j];           // bias rows (k==16, j<8 region)
          else              v = (_Float16)0.0f;  // zero padding
        } else {
          if (s < 16) v = hh[s] * xs[j];
          else        v = xs[j];                 // bias rows fill the whole last step
        }
        A[j] = v;
      }
#pragma unroll
      for (int t = 0; t < NT; ++t) {
        acc[t] = __builtin_amdgcn_wmma_f32_16x16x32_f16(
            false, A, false, B[s][t], (short)0, acc[t], false, false);
      }
    }

    // scatter-add messages: C layout lane = N col (lane&15), VGPR r = M row (half*8+r)
    int dn[8];
#pragma unroll
    for (int r = 0; r < 8; ++r) dn[r] = dst[e0 + half * 8 + r];
#pragma unroll
    for (int t = 0; t < NT; ++t) {
      const int o = t * 16 + arow;
#pragma unroll
      for (int r = 0; r < 8; ++r) {
        atomicAdd(&agg[(size_t)dn[r] * OUT + o], acc[t][r]);
      }
    }
  }
}

// ---------------- node update: h = relu(agg + x @ root + bias), in place ----------------
template <int IN, int OUT>
__global__ void node_update_kernel(float* __restrict__ hio,
                                   const float* __restrict__ xin,
                                   const float* __restrict__ root,
                                   const float* __restrict__ bias) {
  const int idx = blockIdx.x * blockDim.x + threadIdx.x;
  if (idx >= GN_NODES * OUT) return;
  const int n = idx / OUT;
  const int o = idx - n * OUT;
  float s = hio[idx] + bias[o];
#pragma unroll
  for (int i = 0; i < IN; ++i) s = fmaf(xin[n * IN + i], root[i * OUT + o], s);
  hio[idx] = fmaxf(s, 0.0f);
}

// ---------------- mean pool accumulation ----------------
__global__ void pool_kernel(const float* __restrict__ h2, const int* __restrict__ batch,
                            float* __restrict__ msum, float* __restrict__ mcnt) {
  const int n = blockIdx.x * blockDim.x + threadIdx.x;
  if (n >= GN_NODES) return;
  const int b = batch[n];
#pragma unroll
  for (int i = 0; i < 16; ++i) atomicAdd(&msum[b * 16 + i], h2[n * 16 + i]);
  atomicAdd(&mcnt[b], 1.0f);
}

// ---------------- head: normalize pools, concat, lin1 -> lin2 ----------------
__global__ void head_kernel(const float* __restrict__ msum, const float* __restrict__ mcnt,
                            const float* __restrict__ gnum, const float* __restrict__ gsum,
                            const float* __restrict__ l1w, const float* __restrict__ l1b,
                            const float* __restrict__ l2w, const float* __restrict__ l2b,
                            float* __restrict__ out) {
  const int b = blockIdx.x * blockDim.x + threadIdx.x;
  if (b >= GN_GRAPHS) return;
  float feat[32];
  const float c = fmaxf(mcnt[b], 1.0f);
#pragma unroll
  for (int i = 0; i < 16; ++i) feat[i] = msum[b * 16 + i] / c;         // mean(h2)
  const float s = gsum[b];
  const float inv = (s > 0.0f) ? (1.0f / s) : 0.0f;
#pragma unroll
  for (int i = 0; i < 16; ++i) feat[16 + i] = gnum[b * 16 + i] * inv;  // attention pool
  float y = l2b[0];
#pragma unroll
  for (int j = 0; j < 8; ++j) {
    float t = l1b[j];
#pragma unroll
    for (int i = 0; i < 32; ++i) t = fmaf(feat[i], l1w[i * 8 + j], t);
    y = fmaf(t, l2w[j], y);
  }
  out[b] = y;
}

extern "C" void kernel_launch(void* const* d_in, const int* in_sizes, int n_in,
                              void* d_out, int out_size, void* d_ws, size_t ws_size,
                              hipStream_t stream) {
  (void)in_sizes; (void)n_in; (void)out_size; (void)ws_size;
  const float* x_p    = (const float*)d_in[0];
  const float* ea_p   = (const float*)d_in[2];
  const int*   eidx   = (const int*)  d_in[4];
  const int*   batch  = (const int*)  d_in[5];
  const float* nn1_w1 = (const float*)d_in[6];
  const float* nn1_b1 = (const float*)d_in[7];
  const float* nn1_w2 = (const float*)d_in[8];
  const float* nn1_b2 = (const float*)d_in[9];
  const float* root1  = (const float*)d_in[10];
  const float* bias1  = (const float*)d_in[11];
  const float* nn2_w1 = (const float*)d_in[12];
  const float* nn2_b1 = (const float*)d_in[13];
  const float* nn2_w2 = (const float*)d_in[14];
  const float* nn2_b2 = (const float*)d_in[15];
  const float* root2  = (const float*)d_in[16];
  const float* bias2  = (const float*)d_in[17];
  const float* gw1    = (const float*)d_in[18];
  const float* gb1    = (const float*)d_in[19];
  const float* gw2    = (const float*)d_in[20];
  const float* gb2    = (const float*)d_in[21];
  const float* l1w    = (const float*)d_in[22];
  const float* l1b    = (const float*)d_in[23];
  const float* l2w    = (const float*)d_in[24];
  const float* l2b    = (const float*)d_in[25];

  float* ws = (float*)d_ws;
  size_t off = 0;
  float*    agg1 = ws + off; off += (size_t)GN_NODES * 32;   // conv1 agg -> h1 (in place)
  float*    agg2 = ws + off; off += (size_t)GN_NODES * 16;   // conv2 agg -> h2 (in place)
  float*    gsum = ws + off; off += GN_GRAPHS;
  float*    gnum = ws + off; off += (size_t)GN_GRAPHS * 16;
  float*    msum = ws + off; off += (size_t)GN_GRAPHS * 16;
  float*    mcnt = ws + off; off += GN_GRAPHS;
  unsigned* gmax = (unsigned*)(ws + off); off += GN_GRAPHS;  // key(0) < key(any float)
  const int zeron = (int)off;                                // everything above gets zeroed
  float*    gbuf = ws + off; off += GN_NODES;                // gate logits (no init needed)

  const int* srcp = eidx;
  const int* dstp = eidx + GN_EDGES;

  zero_kernel<<<512, 256, 0, stream>>>(ws, zeron);

  gate1_kernel<<<GN_NODES / 256, 256, 0, stream>>>(x_p, gw1, gb1, gw2, gb2, batch, gbuf, gmax);
  gate2_kernel<<<GN_NODES / 256, 256, 0, stream>>>(x_p, batch, gbuf, gmax, gsum, gnum);

  nnconv_edge_kernel<16, 32><<<256, 256, 0, stream>>>(x_p, ea_p, srcp, dstp,
                                                      nn1_w1, nn1_b1, nn1_w2, nn1_b2, agg1);
  node_update_kernel<16, 32><<<(GN_NODES * 32) / 256, 256, 0, stream>>>(agg1, x_p, root1, bias1);

  nnconv_edge_kernel<32, 16><<<256, 256, 0, stream>>>(agg1, ea_p, srcp, dstp,
                                                      nn2_w1, nn2_b1, nn2_w2, nn2_b2, agg2);
  node_update_kernel<32, 16><<<(GN_NODES * 16) / 256, 256, 0, stream>>>(agg2, agg1, root2, bias2);

  pool_kernel<<<GN_NODES / 256, 256, 0, stream>>>(agg2, batch, msum, mcnt);
  head_kernel<<<(GN_GRAPHS + 255) / 256, 256, 0, stream>>>(msum, mcnt, gnum, gsum,
                                                           l1w, l1b, l2w, l2b, (float*)d_out);
}
